// PixelWiseDotProduct_for_summary_80848464380210
// MI455X (gfx1250) — matrix operand order repaired and checked by reading, not compiled, not run on hardware
//
#include <hip/hip_runtime.h>
#include <hip/hip_bf16.h>

typedef __attribute__((ext_vector_type(16))) __bf16 v16bf;
typedef __attribute__((ext_vector_type(8)))  float  v8f;

#define N_B    8
#define C_DIM  128
#define HW_DIM 36864
#define Q_DIM  128
#define NCHUNK 32
#define CHUNK  (HW_DIM / NCHUNK)   /* 1152 */
#define S_ITER 32
#define ITERS  (CHUNK / S_ITER)    /* 36 */

#define LDA_ROW 136   /* shorts per ldsA row [s][c]; 272B, 16B aligned */
#define LDB_ROW 40    /* shorts per ldsB row [c][s];  80B, 16B aligned */

#define ACC_ELEMS (N_B * NCHUNK * 8 * 8 * 8 * 32)  /* 4,194,304 f32 */
#define ML_ELEMS  (N_B * NCHUNK * Q_DIM)           /* 32,768 f32 */

union FragBF {
  v16bf bf;
  unsigned int u[8];
};

__device__ __forceinline__ unsigned short f2bf(float f) {
  unsigned int u = __float_as_uint(f);
  u += 0x7fffu + ((u >> 16) & 1u);          // round-to-nearest-even
  return (unsigned short)(u >> 16);
}
__device__ __forceinline__ unsigned int pack2bf(float lo, float hi) {
  return (unsigned int)f2bf(lo) | ((unsigned int)f2bf(hi) << 16);
}

// ---------------------------------------------------------------------------
// Kernel A: flash-attention partials. One block = (batch n, HW chunk).
// 8 waves; wave w owns q-tile w (16 queries) x full c=128 accumulator.
// ---------------------------------------------------------------------------
__global__ __launch_bounds__(256) void attn_pool_partial(
    const float* __restrict__ x, const float* __restrict__ Km,
    float* __restrict__ accP, float* __restrict__ mP, float* __restrict__ lP)
{
  __shared__ alignas(16) unsigned short ldsA[S_ITER * LDA_ROW];  // [s][c] bf16
  __shared__ alignas(16) unsigned short ldsB[C_DIM * LDB_ROW];   // [c][s] bf16

  const int chunk = blockIdx.x;
  const int n     = blockIdx.y;
  const int tid   = threadIdx.x;
  const int lane  = tid & 31;
  const int w     = tid >> 5;        // q-tile id
  const int lhalf = lane & 16;       // 0 or 16
  const int sl    = lane & 15;

  const float* xn = x + (size_t)n * C_DIM * HW_DIM;

  // ---- preload GEMM1 B fragments (K matrix, reused for the whole stream) ----
  // B 32x16 bf16 layout: lane -> column q, lanes<16 hold K=c 0..15, lanes>=16 K=16..31
  FragBF b1[4];
  {
    const int q = w * 16 + sl;
    const float* kr = Km + ((size_t)n * Q_DIM + q) * C_DIM;
    #pragma unroll
    for (int kc = 0; kc < 4; ++kc) {
      const int cb = kc * 32 + lhalf;
      alignas(16) float kf[16];
      *(float4*)&kf[0]  = *(const float4*)(kr + cb);
      *(float4*)&kf[4]  = *(const float4*)(kr + cb + 4);
      *(float4*)&kf[8]  = *(const float4*)(kr + cb + 8);
      *(float4*)&kf[12] = *(const float4*)(kr + cb + 12);
      #pragma unroll
      for (int j = 0; j < 8; ++j) b1[kc].u[j] = pack2bf(kf[2*j], kf[2*j+1]);
    }
  }

  const v8f zero8 = {0.f,0.f,0.f,0.f,0.f,0.f,0.f,0.f};
  v8f acc[8];                               // D[q=16, c=128] in 8 c-tiles
  #pragma unroll
  for (int ct = 0; ct < 8; ++ct) acc[ct] = zero8;
  float m = -3.0e38f, l = 0.f;

  const int cR = tid >> 1;                  // 0..127 : c row to stage
  const int sg = (tid & 1) * 16;            // which 16-s half of the 32-s tile

  for (int it = 0; it < ITERS; ++it) {
    const int s0 = chunk * CHUNK + it * S_ITER;
    __syncthreads();                        // previous tile fully consumed
    // ---- cooperative stage: global f32 -> bf16 -> LDS (two layouts) ----
    {
      const float* xp = xn + (size_t)cR * HW_DIM + s0 + sg;
      alignas(16) float f[16];
      *(float4*)&f[0]  = *(const float4*)(xp);
      *(float4*)&f[4]  = *(const float4*)(xp + 4);
      *(float4*)&f[8]  = *(const float4*)(xp + 8);
      *(float4*)&f[12] = *(const float4*)(xp + 12);
      unsigned short hs[16];
      #pragma unroll
      for (int j = 0; j < 16; ++j) hs[j] = f2bf(f[j]);
      unsigned int pb[8];
      #pragma unroll
      for (int j = 0; j < 8; ++j)
        pb[j] = (unsigned)hs[2*j] | ((unsigned)hs[2*j+1] << 16);
      // [c][s] copy (GEMM2 B fragments)
      *(uint4*)&ldsB[cR * LDB_ROW + sg]     = make_uint4(pb[0],pb[1],pb[2],pb[3]);
      *(uint4*)&ldsB[cR * LDB_ROW + sg + 8] = make_uint4(pb[4],pb[5],pb[6],pb[7]);
      // [s][c] copy (GEMM1 A fragments)
      #pragma unroll
      for (int j = 0; j < 16; ++j) ldsA[(sg + j) * LDA_ROW + cR] = hs[j];
      // WGP-scope prefetch (locality 3 -> near caches), two tiles ahead to
      // cover HBM latency while this and the next tile are computed.
      if (it + 2 < ITERS) __builtin_prefetch(xp + 2 * S_ITER, 0, 3);
    }
    __syncthreads();

    // ---- GEMM1: S[s,q] over two 16-s tiles, K-dim = c (4 steps of 32) ----
    v8f sc0 = zero8, sc1 = zero8;
    #pragma unroll
    for (int kc = 0; kc < 4; ++kc) {
      const int c0 = kc * 32 + lhalf;            // A layout: +16 c for upper half
      FragBF a0, a1;
      *(uint4*)&a0.u[0] = *(const uint4*)&ldsA[sl * LDA_ROW + c0];
      *(uint4*)&a0.u[4] = *(const uint4*)&ldsA[sl * LDA_ROW + c0 + 8];
      *(uint4*)&a1.u[0] = *(const uint4*)&ldsA[(16 + sl) * LDA_ROW + c0];
      *(uint4*)&a1.u[4] = *(const uint4*)&ldsA[(16 + sl) * LDA_ROW + c0 + 8];
      sc0 = __builtin_amdgcn_wmma_f32_16x16x32_bf16(false, a0.bf, false, b1[kc].bf,
                                                    (short)0, sc0, false, false);
      sc1 = __builtin_amdgcn_wmma_f32_16x16x32_bf16(false, a1.bf, false, b1[kc].bf,
                                                    (short)0, sc1, false, false);
    }

    // ---- online softmax (per q: q = lane&15, replicated in both halves) ----
    float tmax = sc0[0];
    #pragma unroll
    for (int r = 1; r < 8; ++r) tmax = fmaxf(tmax, sc0[r]);
    #pragma unroll
    for (int r = 0; r < 8; ++r) tmax = fmaxf(tmax, sc1[r]);
    tmax = fmaxf(tmax, __shfl_xor(tmax, 16, 32));
    const float mnew = fmaxf(m, tmax);
    const float scl  = __expf(m - mnew);
    float p0[8], p1[8], ps = 0.f;
    #pragma unroll
    for (int r = 0; r < 8; ++r) { p0[r] = __expf(sc0[r] - mnew); ps += p0[r]; }
    #pragma unroll
    for (int r = 0; r < 8; ++r) { p1[r] = __expf(sc1[r] - mnew); ps += p1[r]; }
    ps += __shfl_xor(ps, 16, 32);
    l = l * scl + ps;
    m = mnew;

    // rescale accumulators: broadcast scale[q] onto D-layout rows (q = r + 8*half)
    const int bsrc = (lane >> 1) & 8;
    float rs[8];
    #pragma unroll
    for (int r = 0; r < 8; ++r) rs[r] = __shfl(scl, r + bsrc, 32);
    #pragma unroll
    for (int ct = 0; ct < 8; ++ct)
      #pragma unroll
      for (int r = 0; r < 8; ++r) acc[ct][r] *= rs[r];

    // ---- A2[q, s0..31] from P: pure lane-local repack (C-layout -> A-layout) ----
    FragBF a2;
    #pragma unroll
    for (int j = 0; j < 4; ++j) a2.u[j]     = pack2bf(p0[2*j], p0[2*j+1]);
    #pragma unroll
    for (int j = 0; j < 4; ++j) a2.u[4 + j] = pack2bf(p1[2*j], p1[2*j+1]);

    // ---- GEMM2: acc[q,c] += A2[q,s] * B2[s,c] for 8 c-tiles ----
    #pragma unroll
    for (int ct = 0; ct < 8; ++ct) {
      const int crow = ct * 16 + sl;
      FragBF b2;
      *(uint4*)&b2.u[0] = *(const uint4*)&ldsB[crow * LDB_ROW + (lhalf >> 1)];
      *(uint4*)&b2.u[4] = *(const uint4*)&ldsB[crow * LDB_ROW + (lhalf >> 1) + 16];
      acc[ct] = __builtin_amdgcn_wmma_f32_16x16x32_bf16(false, a2.bf, false, b2.bf,
                                                        (short)0, acc[ct], false, false);
    }
  }

  // ---- write chunk partials (fragment-layout, combine kernel agrees) ----
  const size_t base = (((size_t)n * NCHUNK + chunk) * 8 + w) * (8 * 8 * 32);
  #pragma unroll
  for (int ct = 0; ct < 8; ++ct)
    #pragma unroll
    for (int r = 0; r < 8; ++r)
      accP[base + ((size_t)ct * 8 + r) * 32 + lane] = acc[ct][r];
  if (lane < 16) {
    const size_t mi = ((size_t)n * NCHUNK + chunk) * Q_DIM + w * 16 + lane;
    mP[mi] = m;
    lP[mi] = l;
  }
}

// ---------------------------------------------------------------------------
// Kernel B: merge chunk partials with max/ratio correction, normalize, emit f32.
// block = 256 = 8 r-rows x 32 lanes; grid = (8 q-tiles, 8 batches).
// ---------------------------------------------------------------------------
__global__ __launch_bounds__(256) void attn_pool_combine(
    const float* __restrict__ accP, const float* __restrict__ mP,
    const float* __restrict__ lP, float* __restrict__ out)
{
  const int qt = blockIdx.x, n = blockIdx.y;
  const int t = threadIdx.x;
  const int lane = t & 31;
  const int r = t >> 5;
  const int q = qt * 16 + r + ((lane & 16) >> 1);   // D-layout row -> q

  float M = -3.0e38f;
  for (int ch = 0; ch < NCHUNK; ++ch)
    M = fmaxf(M, mP[((size_t)n * NCHUNK + ch) * Q_DIM + q]);
  float wgt[NCHUNK];
  float L = 0.f;
  for (int ch = 0; ch < NCHUNK; ++ch) {
    const size_t mi = ((size_t)n * NCHUNK + ch) * Q_DIM + q;
    const float e = __expf(mP[mi] - M);
    wgt[ch] = e;
    L += e * lP[mi];
  }
  const float invL = 1.0f / L;
  #pragma unroll
  for (int ct = 0; ct < 8; ++ct) {
    float s = 0.f;
    for (int ch = 0; ch < NCHUNK; ++ch) {
      const size_t ai = ((((size_t)n * NCHUNK + ch) * 8 + qt) * (8 * 8 * 32))
                      + ((size_t)ct * 8 + r) * 32 + lane;
      s += wgt[ch] * accP[ai];
    }
    const int c = ct * 16 + (lane & 15);
    out[((size_t)n * Q_DIM + q) * C_DIM + c] = s * invL;
  }
}

extern "C" void kernel_launch(void* const* d_in, const int* in_sizes, int n_in,
                              void* d_out, int out_size, void* d_ws, size_t ws_size,
                              hipStream_t stream) {
  (void)in_sizes; (void)n_in; (void)out_size; (void)ws_size;
  const float* x  = (const float*)d_in[0];
  const float* Km = (const float*)d_in[1];
  float* accP = (float*)d_ws;                 // 16 MB of partial accumulators
  float* mP   = accP + ACC_ELEMS;
  float* lP   = mP + ML_ELEMS;
  attn_pool_partial<<<dim3(NCHUNK, N_B), dim3(256), 0, stream>>>(x, Km, accP, mP, lP);
  attn_pool_combine<<<dim3(8, N_B), dim3(256), 0, stream>>>(accP, mP, lP, (float*)d_out);
}
